// DefiNet_58643483459788
// MI455X (gfx1250) — matrix operand order repaired and checked by reading, not compiled, not run on hardware
//
#include <hip/hip_runtime.h>
#include <hip/hip_bf16.h>

#define N_NODES 50000
#define N_EDGES 800000
#define N_GRAPHS 256
#define D 64
#define ORIG 92
#define NBR 41
#define NCONV 4
#define H 128
#define NMARK 2
#define KZ 192                 // padded 2*D + NBR
#define LN2F 0.6931471805599453f
#define NODE_BLOCKS ((N_NODES + 31) / 32)      // 1563
#define N_PAD (NODE_BLOCKS * 32)               // 50016 (padded atom rows)

typedef __attribute__((ext_vector_type(16))) _Float16 v16h;
typedef __attribute__((ext_vector_type(8)))  _Float16 v8h;
typedef __attribute__((ext_vector_type(8)))  float    v8f;

// ---------------- math helpers ----------------
__device__ __forceinline__ float softplusf_(float x) {
    return (x > 20.f) ? x : __logf(1.f + __expf(x));
}
__device__ __forceinline__ float sspf_(float x) {       // shifted softplus
    return softplusf_(x) - LN2F;
}
__device__ __forceinline__ float sigmf_(float x) {
    return 1.f / (1.f + __expf(-x));
}
__device__ __forceinline__ v8h cvt8(float4 a, float4 b) {
    v8h r;
    r[0] = (_Float16)a.x; r[1] = (_Float16)a.y; r[2] = (_Float16)a.z; r[3] = (_Float16)a.w;
    r[4] = (_Float16)b.x; r[5] = (_Float16)b.y; r[6] = (_Float16)b.z; r[7] = (_Float16)b.w;
    return r;
}

// ---------------- WMMA helpers ----------------
// A fragment: 16x32 f16 tile from LDS (row-major, strideH halves). ISA 7.12.2 layout.
__device__ __forceinline__ v16h load_a_frag(const _Float16* lds, int row0,
                                            int strideH, int k0, int lane) {
    int m    = lane & 15;
    int koff = (lane & 16) ? 8 : 0;
    const _Float16* p = lds + (size_t)(row0 + m) * strideH + k0 + koff;
    union { v16h v; v8h h[2]; } u;
    u.h[0] = *(const v8h*)(p);
    u.h[1] = *(const v8h*)(p + 16);
    return u.v;
}
// B fragment: pre-repacked weights, fragment-contiguous.
__device__ __forceinline__ v16h load_b_frag(const _Float16* Wp, int frag, int lane) {
    const _Float16* p = Wp + ((size_t)frag * 32 + lane) * 16;
    union { v16h v; v8h h[2]; } u;
    u.h[0] = *(const v8h*)(p);
    u.h[1] = *(const v8h*)(p + 8);
    return u.v;
}
__device__ __forceinline__ v8f wmma16(v16h a, v16h b, v8f c) {
    return __builtin_amdgcn_wmma_f32_16x16x32_f16(false, a, false, b,
                                                  (short)0, c, false, false);
}

// ---------------- weight repack: f32 [K x N] row-major -> f16 B fragments ----------------
__global__ void repack_b_kernel(const float* __restrict__ W, _Float16* __restrict__ out,
                                int K, int N, int ksteps) {
    int idx = blockIdx.x * blockDim.x + threadIdx.x;
    int ntiles = N / 16;
    int total  = ksteps * ntiles * 32;
    if (idx >= total) return;
    int lane = idx & 31;
    int f    = idx >> 5;
    int nt   = f % ntiles;
    int kk   = f / ntiles;
    int n     = nt * 16 + (lane & 15);
    int kbase = kk * 32 + ((lane & 16) ? 8 : 0);
    _Float16* dst = out + ((size_t)f * 32 + lane) * 16;
#pragma unroll
    for (int h = 0; h < 16; ++h) {
        int k = kbase + (h < 8 ? h : h + 8);
        float v = (k < K) ? W[(size_t)k * N + n] : 0.f;
        dst[h] = (_Float16)v;
    }
}

// ---------------- misc elementwise ----------------
__global__ void zero_kernel(float* __restrict__ p, long n) {
    long i = (long)blockIdx.x * blockDim.x + threadIdx.x;
    if (i < n) p[i] = 0.f;
}
__global__ void atomh_kernel(const float* __restrict__ a, _Float16* __restrict__ ah) {
    long i = (long)blockIdx.x * blockDim.x + threadIdx.x;
    if (i < (long)N_NODES * D) ah[i] = (_Float16)a[i];
}
__global__ void ginit_kernel(float* __restrict__ g, const float* __restrict__ seed) {
    int i = blockIdx.x * blockDim.x + threadIdx.x;
    if (i < N_GRAPHS * D) g[i] = seed[i & (D - 1)];
}

// ---------------- embedding: atom = x @ emb_W + emb_b (WMMA, K=92->96) ----------------
__global__ __launch_bounds__(256) void embed_kernel(const float* __restrict__ x,
                                                    const _Float16* __restrict__ Wp,
                                                    const float* __restrict__ b,
                                                    float* __restrict__ atom) {
    __shared__ __align__(16) _Float16 zin[32 * 96];
    int t = threadIdx.x;
    long r0 = (long)blockIdx.x * 32;
    bool full = (r0 + 32 <= N_NODES);
    if (full) {
        for (int idx = t; idx < 32 * 96; idx += 256) {
            int r = idx / 96, c = idx % 96;
            zin[idx] = (c < ORIG) ? (_Float16)x[(r0 + r) * ORIG + c] : (_Float16)0.f;
        }
    } else {
        for (int idx = t; idx < 32 * 96; idx += 256) {
            int r = idx / 96, c = idx % 96;
            long node = r0 + r;
            float v = (node < N_NODES && c < ORIG) ? x[node * ORIG + c] : 0.f;
            zin[idx] = (_Float16)v;
        }
    }
    __syncthreads();
    int wave = t >> 5, lane = t & 31;
    int rt = wave >> 2, nt = wave & 3;
    v8f acc = {};
#pragma unroll
    for (int kk = 0; kk < 3; ++kk) {
        v16h a  = load_a_frag(zin, rt * 16, 96, kk * 32, lane);
        v16h bb = load_b_frag(Wp, kk * 4 + nt, lane);
        acc = wmma16(a, bb, acc);
    }
    union { v8f v; float f[8]; } u; u.v = acc;
    int n = nt * 16 + (lane & 15);
    float bv = b[n];
    int msel = (lane & 16) ? 8 : 0;
#pragma unroll
    for (int v = 0; v < 8; ++v) {            // atom is padded: no bounds check
        long node = r0 + rt * 16 + v + msel;
        atom[node * D + n] = u.f[v] + bv;
    }
}

// ---------------- dist MLP: atom += W2( ssp(W1([atom, g[batch]])+b1) )+b2 ----------------
__global__ __launch_bounds__(256) void dist_kernel(float* __restrict__ atom,
                                                   const float* __restrict__ g,
                                                   const int* __restrict__ batch,
                                                   const _Float16* __restrict__ W1p,
                                                   const float* __restrict__ b1,
                                                   const _Float16* __restrict__ W2p,
                                                   const float* __restrict__ b2) {
    __shared__ __align__(16) _Float16 zin[32 * 128];
    __shared__ __align__(16) _Float16 hid[32 * 64];
    int t = threadIdx.x;
    long r0 = (long)blockIdx.x * 32;
    bool full = (r0 + 32 <= N_NODES);
    if (full) {
        // vectorized fill: thread t -> row t>>3, 8-float chunk (t&7)
        int r = t >> 3, c8 = (t & 7) * 8;
        long node = r0 + r;
        const float4* pa = (const float4*)(atom + node * D + c8);
        float4 a0 = pa[0], a1 = pa[1];
        *(v8h*)(&zin[r * 128 + c8]) = cvt8(a0, a1);
        const float4* pg = (const float4*)(g + (size_t)batch[node] * D + c8);
        float4 g0 = pg[0], g1 = pg[1];
        *(v8h*)(&zin[r * 128 + 64 + c8]) = cvt8(g0, g1);
    } else {
        for (int idx = t; idx < 32 * 128; idx += 256) {
            int r = idx >> 7, c = idx & 127;
            long node = r0 + r;
            float v = 0.f;
            if (node < N_NODES)
                v = (c < D) ? atom[node * D + c] : g[(size_t)batch[node] * D + (c - D)];
            zin[idx] = (_Float16)v;
        }
    }
    __syncthreads();
    int wave = t >> 5, lane = t & 31;
    int rt = wave >> 2, nt = wave & 3;
    int n = nt * 16 + (lane & 15);
    int msel = (lane & 16) ? 8 : 0;

    v8f acc = {};
#pragma unroll
    for (int kk = 0; kk < 4; ++kk) {
        v16h a  = load_a_frag(zin, rt * 16, 128, kk * 32, lane);
        v16h bb = load_b_frag(W1p, kk * 4 + nt, lane);
        acc = wmma16(a, bb, acc);
    }
    union { v8f v; float f[8]; } u; u.v = acc;
    float b1v = b1[n];
#pragma unroll
    for (int v = 0; v < 8; ++v) {
        int m = rt * 16 + v + msel;
        hid[m * 64 + n] = (_Float16)sspf_(u.f[v] + b1v);
    }
    __syncthreads();

    v8f acc2 = {};
#pragma unroll
    for (int kk = 0; kk < 2; ++kk) {
        v16h a  = load_a_frag(hid, rt * 16, 64, kk * 32, lane);
        v16h bb = load_b_frag(W2p, kk * 4 + nt, lane);
        acc2 = wmma16(a, bb, acc2);
    }
    union { v8f v; float f[8]; } u2; u2.v = acc2;
    float b2v = b2[n];
    if (full) {
#pragma unroll
        for (int v = 0; v < 8; ++v) {
            long node = r0 + rt * 16 + v + msel;
            atom[node * D + n] += u2.f[v] + b2v;
        }
    } else {
#pragma unroll
        for (int v = 0; v < 8; ++v) {
            long node = r0 + rt * 16 + v + msel;
            if (node < N_NODES) atom[node * D + n] += u2.f[v] + b2v;
        }
    }
}

// ---------------- edge conv: gate*core -> atomic scatter to aggr ----------------
// N_EDGES is an exact multiple of 32, so every block takes the fast path; the
// guarded path is kept for generality.
__global__ __launch_bounds__(256) void conv_edge_kernel(const _Float16* __restrict__ atom_h,
                                                        const float* __restrict__ edge_attr,
                                                        const int* __restrict__ esrc,
                                                        const int* __restrict__ edst,
                                                        const int* __restrict__ marker,
                                                        const _Float16* __restrict__ Wfp,
                                                        const _Float16* __restrict__ Wsp,
                                                        const float* __restrict__ bf,
                                                        const float* __restrict__ bs,
                                                        const float* __restrict__ pemb,
                                                        float* __restrict__ aggr) {
    __shared__ __align__(16) _Float16 z[32 * KZ];
    __shared__ int dstv[32];
    __shared__ int srcv[32];
    __shared__ int pairv[32];
    int t = threadIdx.x;
    long e0 = (long)blockIdx.x * 32;
    bool full = (e0 + 32 <= N_EDGES);
    if (t < 32) {
        long e = e0 + t;
        if (e < N_EDGES) {
            int s = esrc[e], d0 = edst[e];
            dstv[t] = d0; srcv[t] = s;
            pairv[t] = marker[d0] * NMARK + marker[s];
        } else { dstv[t] = -1; srcv[t] = 0; pairv[t] = 0; }
    }
    __syncthreads();
    if (full) {
        // async gather of atom rows into LDS: per-lane 16B chunks
        int r = t >> 3, c = t & 7;                       // row 0..31, chunk 0..7
        unsigned ldsd = (unsigned)(uintptr_t)(&z[r * KZ + c * 8]);
        unsigned ldss = (unsigned)(uintptr_t)(&z[r * KZ + D + c * 8]);
        const _Float16* gd = atom_h + (size_t)dstv[r] * D + c * 8;
        const _Float16* gs = atom_h + (size_t)srcv[r] * D + c * 8;
        asm volatile("global_load_async_to_lds_b128 %0, %1, off"
                     :: "v"(ldsd), "v"(gd) : "memory");
        asm volatile("global_load_async_to_lds_b128 %0, %1, off"
                     :: "v"(ldss), "v"(gs) : "memory");
        // edge_attr segment (f32 -> f16 with zero pad to 64)
        for (int idx = t; idx < 32 * 64; idx += 256) {
            int rr = idx >> 6, cc = idx & 63;
            _Float16 val = (_Float16)0.f;
            if (cc < NBR) val = (_Float16)edge_attr[(size_t)(e0 + rr) * NBR + cc];
            z[rr * KZ + 2 * D + cc] = val;
        }
        asm volatile("s_wait_asynccnt 0x0" ::: "memory");
    } else {
        for (int idx = t; idx < 32 * KZ; idx += 256) {
            int r = idx / KZ, c = idx % KZ;
            long e = e0 + r;
            _Float16 val = (_Float16)0.f;
            if (e < N_EDGES) {
                if (c < D)                 val = atom_h[(size_t)dstv[r] * D + c];
                else if (c < 2 * D)        val = atom_h[(size_t)srcv[r] * D + (c - D)];
                else if (c < 2 * D + NBR)  val = (_Float16)edge_attr[(size_t)e * NBR + (c - 2 * D)];
            }
            z[idx] = val;
        }
    }
    __syncthreads();
    int wave = t >> 5, lane = t & 31;
    int rt = wave >> 2, nt = wave & 3;
    v8f accf = {}, accs = {};
#pragma unroll
    for (int kk = 0; kk < 6; ++kk) {
        v16h a   = load_a_frag(z, rt * 16, KZ, kk * 32, lane);
        v16h bfr = load_b_frag(Wfp, kk * 4 + nt, lane);
        v16h bsr = load_b_frag(Wsp, kk * 4 + nt, lane);
        accf = wmma16(a, bfr, accf);
        accs = wmma16(a, bsr, accs);
    }
    union { v8f v; float f[8]; } uf, us;
    uf.v = accf; us.v = accs;
    int n = nt * 16 + (lane & 15);
    float bfv = bf[n], bsv = bs[n];
    float pe0 = pemb[0 * D + n], pe1 = pemb[1 * D + n];
    float pe2 = pemb[2 * D + n], pe3 = pemb[3 * D + n];
    int msel = (lane & 16) ? 8 : 0;
#pragma unroll
    for (int v = 0; v < 8; ++v) {
        int m = rt * 16 + v + msel;
        int d0 = dstv[m];
        int pp = pairv[m];
        float pv = (pp == 0) ? pe0 : (pp == 1) ? pe1 : (pp == 2) ? pe2 : pe3;
        float gate = sigmf_(uf.f[v] + bfv + pv);
        float core = sspf_(us.f[v] + bsv);
        if (full) {
            atomicAdd(&aggr[(size_t)d0 * D + n], gate * core);
        } else if (d0 >= 0) {
            atomicAdd(&aggr[(size_t)d0 * D + n], gate * core);
        }
    }
}

// ---------------- BN stats / apply ----------------
__global__ void bnstats_kernel(const float* __restrict__ aggr, float* __restrict__ stats) {
    int f = threadIdx.x & 63;
    int rlane = threadIdx.x >> 6;
    float s = 0.f, ss = 0.f;
    for (long r = (long)blockIdx.x * 4 + rlane; r < N_NODES; r += (long)gridDim.x * 4) {
        float v = aggr[r * D + f];
        s += v; ss += v * v;
    }
    atomicAdd(&stats[f], s);
    atomicAdd(&stats[D + f], ss);
}
__global__ void bnapply_kernel(float* __restrict__ atom, const float* __restrict__ aggr,
                               const float* __restrict__ stats,
                               const float* __restrict__ gamma, const float* __restrict__ beta) {
    long i = (long)blockIdx.x * blockDim.x + threadIdx.x;
    if (i >= (long)N_NODES * D) return;
    int f = (int)(i & 63);
    const float inv = 1.f / (float)N_NODES;
    float mu  = stats[f] * inv;
    float var = stats[D + f] * inv - mu * mu;
    atom[i] += (aggr[i] - mu) * rsqrtf(var + 1e-5f) * gamma[f] + beta[f];
}

// ---------------- pooling (segment mean over batch) ----------------
__global__ void pool_kernel(const float* __restrict__ atom, const int* __restrict__ batch,
                            float* __restrict__ pooled, float* __restrict__ counts) {
    long i = (long)blockIdx.x * blockDim.x + threadIdx.x;
    if (i >= (long)N_NODES * D) return;
    long node = i >> 6;
    int f = (int)(i & 63);
    int b = batch[node];
    atomicAdd(&pooled[(size_t)b * D + f], atom[i]);
    if (f == 0) atomicAdd(&counts[b], 1.f);
}
__global__ void pooldiv_kernel(float* __restrict__ pooled, const float* __restrict__ counts) {
    int i = blockIdx.x * blockDim.x + threadIdx.x;
    if (i >= N_GRAPHS * D) return;
    pooled[i] /= fmaxf(counts[i >> 6], 1.f);
}

// ---------------- global-state MLP (G=256, scalar) ----------------
__global__ void gmlp1_kernel(const float* __restrict__ pooled, const float* __restrict__ g,
                             const float* __restrict__ W1, const float* __restrict__ b1,
                             float* __restrict__ hid) {
    int i = blockIdx.x * blockDim.x + threadIdx.x;
    if (i >= N_GRAPHS * D) return;
    int r = i >> 6, n = i & 63;
    float acc = b1[n];
    for (int k = 0; k < D; ++k) acc += pooled[r * D + k] * W1[k * D + n];
    for (int k = 0; k < D; ++k) acc += g[r * D + k] * W1[(D + k) * D + n];
    hid[i] = sspf_(acc);
}
__global__ void gmlp2_kernel(float* __restrict__ g, const float* __restrict__ hid,
                             const float* __restrict__ W2, const float* __restrict__ b2) {
    int i = blockIdx.x * blockDim.x + threadIdx.x;
    if (i >= N_GRAPHS * D) return;
    int r = i >> 6, n = i & 63;
    float acc = b2[n];
    for (int k = 0; k < D; ++k) acc += hid[r * D + k] * W2[k * D + n];
    g[i] += acc;
}

// ---------------- head ----------------
__global__ void crys_kernel(const float* __restrict__ pooled, const float* __restrict__ g,
                            float* __restrict__ crys) {
    int i = blockIdx.x * blockDim.x + threadIdx.x;
    if (i >= N_GRAPHS * 2 * D) return;
    int r = i >> 7, c = i & 127;
    float v = (c < D) ? pooled[r * D + c] : g[r * D + (c - D)];
    crys[i] = softplusf_(v);
}
__global__ void fc_kernel(const float* __restrict__ crys, const float* __restrict__ W,
                          const float* __restrict__ b, float* __restrict__ h) {
    int i = blockIdx.x * blockDim.x + threadIdx.x;
    if (i >= N_GRAPHS * H) return;
    int r = i >> 7, n = i & 127;
    float acc = b[n];
    for (int k = 0; k < 2 * D; ++k) acc += crys[r * 2 * D + k] * W[k * H + n];
    h[i] = softplusf_(acc);
}
__global__ void out_kernel(const float* __restrict__ h, const float* __restrict__ W,
                           const float* __restrict__ b, float* __restrict__ out) {
    int r = blockIdx.x * blockDim.x + threadIdx.x;
    if (r >= N_GRAPHS) return;
    float acc = b[0];
    for (int k = 0; k < H; ++k) acc += h[r * H + k] * W[k];
    out[r] = acc;
}

// =====================================================================
extern "C" void kernel_launch(void* const* d_in, const int* in_sizes, int n_in,
                              void* d_out, int out_size, void* d_ws, size_t ws_size,
                              hipStream_t stream) {
    const float* x          = (const float*)d_in[0];
    const float* edge_attr  = (const float*)d_in[1];
    const int*   edge_index = (const int*)d_in[2];
    const int*   batch      = (const int*)d_in[3];
    const int*   marker     = (const int*)d_in[4];
    const float* emb_W      = (const float*)d_in[5];
    const float* emb_b      = (const float*)d_in[6];
    const float* gseed      = (const float*)d_in[7];
    const float* dist_W1    = (const float*)d_in[8];
    const float* dist_b1    = (const float*)d_in[9];
    const float* dist_W2    = (const float*)d_in[10];
    const float* dist_b2    = (const float*)d_in[11];
    const float* conv_Wf    = (const float*)d_in[12];
    const float* conv_bf    = (const float*)d_in[13];
    const float* conv_Ws    = (const float*)d_in[14];
    const float* conv_bs    = (const float*)d_in[15];
    const float* pair_emb   = (const float*)d_in[16];
    const float* bn_gamma   = (const float*)d_in[17];
    const float* bn_beta    = (const float*)d_in[18];
    const float* agg_W1     = (const float*)d_in[19];
    const float* agg_b1     = (const float*)d_in[20];
    const float* agg_W2     = (const float*)d_in[21];
    const float* agg_b2     = (const float*)d_in[22];
    const float* fc_W       = (const float*)d_in[23];
    const float* fc_b       = (const float*)d_in[24];
    const float* out_W      = (const float*)d_in[25];
    const float* out_b      = (const float*)d_in[26];

    const int* esrc = edge_index;
    const int* edst = edge_index + N_EDGES;

    // ---- workspace carve-up (256B aligned regions) ----
    char* wptr = (char*)d_ws;
    auto alloc = [&](size_t bytes) -> void* {
        void* p = (void*)wptr;
        wptr += (bytes + 255) & ~((size_t)255);
        return p;
    };
    float*     atom   = (float*)alloc((size_t)N_PAD * D * 4);      // padded rows
    _Float16*  atom_h = (_Float16*)alloc((size_t)N_NODES * D * 2);
    float*     aggr   = (float*)alloc((size_t)N_NODES * D * 4);
    float*     g      = (float*)alloc((size_t)N_GRAPHS * D * 4);
    float*     pooled = (float*)alloc((size_t)N_GRAPHS * D * 4);
    float*     counts = (float*)alloc((size_t)N_GRAPHS * 4);
    float*     stats  = (float*)alloc((size_t)2 * D * 4);
    float*     hidg   = (float*)alloc((size_t)N_GRAPHS * D * 4);
    float*     crys   = (float*)alloc((size_t)N_GRAPHS * 2 * D * 4);
    float*     fch    = (float*)alloc((size_t)N_GRAPHS * H * 4);
    _Float16*  embWp  = (_Float16*)alloc((size_t)3 * 4 * 512 * 2);
    _Float16*  dW1p   = (_Float16*)alloc((size_t)NCONV * 16 * 512 * 2);
    _Float16*  dW2p   = (_Float16*)alloc((size_t)NCONV * 8  * 512 * 2);
    _Float16*  cWfp   = (_Float16*)alloc((size_t)NCONV * 24 * 512 * 2);
    _Float16*  cWsp   = (_Float16*)alloc((size_t)NCONV * 24 * 512 * 2);

    const int TPB = 256;
    const int edgeBlocks = (N_EDGES + 31) / 32;                // 25000
    const long ndElems   = (long)N_NODES * D;                  // 3.2M
    const int ndBlocks   = (int)((ndElems + TPB - 1) / TPB);   // 12500
    const int gdBlocks   = (N_GRAPHS * D + TPB - 1) / TPB;

    // ---- repack all weight matrices into WMMA B-fragment layout ----
    repack_b_kernel<<<2, TPB, 0, stream>>>(emb_W, embWp, ORIG, D, 3);
    for (int l = 0; l < NCONV; ++l) {
        repack_b_kernel<<<2, TPB, 0, stream>>>(dist_W1 + (size_t)l * 2 * D * D, dW1p + (size_t)l * 16 * 512, 2 * D, D, 4);
        repack_b_kernel<<<1, TPB, 0, stream>>>(dist_W2 + (size_t)l * D * D,     dW2p + (size_t)l * 8  * 512, D, D, 2);
        repack_b_kernel<<<3, TPB, 0, stream>>>(conv_Wf + (size_t)l * (2 * D + NBR) * D, cWfp + (size_t)l * 24 * 512, 2 * D + NBR, D, 6);
        repack_b_kernel<<<3, TPB, 0, stream>>>(conv_Ws + (size_t)l * (2 * D + NBR) * D, cWsp + (size_t)l * 24 * 512, 2 * D + NBR, D, 6);
    }

    // ---- embedding + global-state init ----
    embed_kernel<<<NODE_BLOCKS, TPB, 0, stream>>>(x, embWp, emb_b, atom);
    ginit_kernel<<<gdBlocks, TPB, 0, stream>>>(g, gseed);

    for (int l = 0; l < NCONV; ++l) {
        dist_kernel<<<NODE_BLOCKS, TPB, 0, stream>>>(atom, g, batch,
                                                     dW1p + (size_t)l * 16 * 512, dist_b1 + l * D,
                                                     dW2p + (size_t)l * 8 * 512,  dist_b2 + l * D);
        atomh_kernel<<<ndBlocks, TPB, 0, stream>>>(atom, atom_h);
        zero_kernel<<<ndBlocks, TPB, 0, stream>>>(aggr, ndElems);
        zero_kernel<<<1, 2 * D, 0, stream>>>(stats, 2 * D);
        conv_edge_kernel<<<edgeBlocks, TPB, 0, stream>>>(atom_h, edge_attr, esrc, edst, marker,
                                                         cWfp + (size_t)l * 24 * 512,
                                                         cWsp + (size_t)l * 24 * 512,
                                                         conv_bf + l * D, conv_bs + l * D,
                                                         pair_emb + (size_t)l * NMARK * NMARK * D,
                                                         aggr);
        bnstats_kernel<<<256, TPB, 0, stream>>>(aggr, stats);
        bnapply_kernel<<<ndBlocks, TPB, 0, stream>>>(atom, aggr, stats,
                                                     bn_gamma + l * D, bn_beta + l * D);
        zero_kernel<<<gdBlocks, TPB, 0, stream>>>(pooled, N_GRAPHS * D);
        zero_kernel<<<1, N_GRAPHS, 0, stream>>>(counts, N_GRAPHS);
        pool_kernel<<<ndBlocks, TPB, 0, stream>>>(atom, batch, pooled, counts);
        pooldiv_kernel<<<gdBlocks, TPB, 0, stream>>>(pooled, counts);
        gmlp1_kernel<<<gdBlocks, TPB, 0, stream>>>(pooled, g,
                                                   agg_W1 + (size_t)l * 2 * D * D,
                                                   agg_b1 + l * D, hidg);
        gmlp2_kernel<<<gdBlocks, TPB, 0, stream>>>(g, hidg,
                                                   agg_W2 + (size_t)l * D * D,
                                                   agg_b2 + l * D);
    }

    // ---- head ----
    zero_kernel<<<gdBlocks, TPB, 0, stream>>>(pooled, N_GRAPHS * D);
    zero_kernel<<<1, N_GRAPHS, 0, stream>>>(counts, N_GRAPHS);
    pool_kernel<<<ndBlocks, TPB, 0, stream>>>(atom, batch, pooled, counts);
    pooldiv_kernel<<<gdBlocks, TPB, 0, stream>>>(pooled, counts);
    crys_kernel<<<(N_GRAPHS * 2 * D + TPB - 1) / TPB, TPB, 0, stream>>>(pooled, g, crys);
    fc_kernel<<<(N_GRAPHS * H + TPB - 1) / TPB, TPB, 0, stream>>>(crys, fc_W, fc_b, fch);
    out_kernel<<<1, N_GRAPHS, 0, stream>>>(fch, out_W, out_b, (float*)d_out);
}
